// ActionNet_64854006169605
// MI455X (gfx1250) — matrix operand order repaired and checked by reading, not compiled, not run on hardware
//
#include <hip/hip_runtime.h>
#include <hip/hip_bf16.h>
#include <stdint.h>

typedef unsigned short u16;
typedef __attribute__((ext_vector_type(16))) __bf16 v16bf;
typedef __attribute__((ext_vector_type(8)))  float  v8f;

// ---------------------------------------------------------------- utilities
__device__ inline u16 f2bf(float f) {
    union { float f; unsigned u; } x; x.f = f;
    unsigned r = x.u + 0x7FFFu + ((x.u >> 16) & 1u);   // round-to-nearest-even
    return (u16)(r >> 16);
}

union U16bf { v16bf v; uint4 q[2]; u16 h[16]; };

// A-fragment (16x32 bf16): lane lo = M row, elements 0..7 -> K=hi*8+e,
// elements 8..15 -> K=16+hi*8+(e-8).  LDS row stride = 40 u16 (16B aligned).
__device__ inline v16bf load_frag_a(const u16* lds, int mt, int lane) {
    int lo = lane & 15, hi = lane >> 4;
    const u16* row = lds + (mt * 16 + lo) * 40;
    U16bf u;
    u.q[0] = *(const uint4*)(row + hi * 8);
    u.q[1] = *(const uint4*)(row + 16 + hi * 8);
    return u.v;
}

// B-fragment (32x16 bf16), staged N-major in LDS: lane lo = N column,
// elements 0..7 -> K=hi*8+e, elements 8..15 -> K=16+hi*8+(e-8).
__device__ inline v16bf load_frag_b(const u16* lds, int nt, int lane) {
    int lo = lane & 15, hi = lane >> 4;
    const u16* row = lds + (nt * 16 + lo) * 40;
    U16bf u;
    u.q[0] = *(const uint4*)(row + hi * 8);
    u.q[1] = *(const uint4*)(row + 16 + hi * 8);
    return u.v;
}

__device__ inline v8f wmma_bf16(v16bf a, v16bf b, v8f c) {
    return __builtin_amdgcn_wmma_f32_16x16x32_bf16(false, a, false, b, (short)0, c, false, false);
}

__device__ inline v8f vzero8() { v8f z; for (int i = 0; i < 8; ++i) z[i] = 0.f; return z; }

// ---------------------------------------------- CDNA5 async global->LDS path
// GLOBAL_LOAD_ASYNC_TO_LDS_B128: per-lane 16B transfer into LDS, ASYNCcnt.
// LDS byte address = low 32 bits of the generic pointer (aperture truncation).
__device__ inline void async_ld_b128(const void* gsrc, void* lds_dst) {
    unsigned l = (unsigned)(uintptr_t)lds_dst;
    asm volatile("global_load_async_to_lds_b128 %0, %1, off"
                 :: "v"(l), "v"(gsrc) : "memory");
}

template <int N> __device__ inline void s_wait_async() {
    static_assert(N == 0 || N == 2 || N == 3, "unsupported asynccnt wait");
    if constexpr (N == 0)      asm volatile("s_wait_asynccnt 0" ::: "memory");
    else if constexpr (N == 2) asm volatile("s_wait_asynccnt 2" ::: "memory");
    else                       asm volatile("s_wait_asynccnt 3" ::: "memory");
}

// ------------------------------------------------- Wigner J_l(pi/2) blocks
__device__ double dfact(int n) { double r = 1.0; for (int i = 2; i <= n; ++i) r *= i; return r; }
__device__ double ipow(double b, int e) { double r = 1.0; for (int i = 0; i < e; ++i) r *= b; return r; }

__constant__ int JDOFF[6] = {0, 1, 10, 35, 84, 165};

__global__ void k_init_jd(float* jd) {
    if (threadIdx.x != 0 || blockIdx.x != 0) return;
    const double ch = 0.70710678118654752440, sh = 0.70710678118654752440;
    for (int l = 0; l <= 5; ++l) {
        const int s = 2 * l + 1;
        double d[11][11], Cre[11][11], Cim[11][11];
        for (int i = 0; i < s; ++i) {
            int mp = i - l;
            for (int j = 0; j < s; ++j) {
                int m = j - l;
                double pref = sqrt(dfact(l + mp) * dfact(l - mp) * dfact(l + m) * dfact(l - m));
                int s0 = (m - mp) > 0 ? (m - mp) : 0;
                int s1 = (l + m) < (l - mp) ? (l + m) : (l - mp);
                double tot = 0.0;
                for (int ss = s0; ss <= s1; ++ss) {
                    double sgn = ((mp - m + ss) & 1) ? -1.0 : 1.0;
                    tot += sgn / (dfact(l + m - ss) * dfact(ss) * dfact(mp - m + ss) * dfact(l - mp - ss))
                           * ipow(ch, 2 * l + m - mp - 2 * ss) * ipow(sh, mp - m + 2 * ss);
                }
                d[i][j] = pref * tot;
            }
        }
        for (int i = 0; i < s; ++i)
            for (int j = 0; j < s; ++j) { Cre[i][j] = 0.0; Cim[i][j] = 0.0; }
        const double is2 = 0.70710678118654752440;
        for (int i = 0; i < s; ++i) {
            int m = i - l;
            if (m < 0) {
                Cim[i][l + m] = is2;
                Cim[i][l - m] = -((m & 1) ? -1.0 : 1.0) * is2;
            } else if (m == 0) {
                Cre[i][l] = 1.0;
            } else {
                Cre[i][l - m] = is2;
                Cre[i][l + m] = ((m & 1) ? -1.0 : 1.0) * is2;
            }
        }
        float* J = jd + JDOFF[l];
        for (int a = 0; a < s; ++a)
            for (int b = 0; b < s; ++b) {
                double acc = 0.0;
                for (int p = 0; p < s; ++p)
                    for (int q = 0; q < s; ++q)
                        acc += d[p][q] * (Cre[a][p] * Cre[b][q] + Cim[a][p] * Cim[b][q]);
                J[a * s + b] = (float)acc;
            }
    }
}

// ------------------------------------------------------------ fill / prepack
__global__ void k_fill_u16(u16* p, int count, u16 val) {
    int i = blockIdx.x * blockDim.x + threadIdx.x;
    if (i < count) p[i] = val;
}

// w0 [360,16384] f32  ->  w0T bf16 [16384, 384] (K padded, N-major)
__global__ void k_prep_w0(const float* w0, u16* w0T) {
    int i = blockIdx.x * blockDim.x + threadIdx.x;
    if (i >= 16384 * 384) return;
    int j = i / 384, k = i % 384;
    float v = (k < 360) ? w0[(size_t)k * 16384 + j] : 0.f;
    w0T[i] = f2bf(v);
}

// conv-transpose tap tables (stride 2, kernel 4, SAME -> pad_lo = 1):
// parity 0: (di,k) in {(0,1),(-1,3)} ; parity 1: {(0,2),(+1,0)}
__device__ inline int tap_k (int par, int t) { return par ? (t ? 0 : 2) : (t ? 3 : 1); }
__device__ inline int tap_di(int par, int t) { return t ? (par ? 1 : -1) : 0; }

// w [COUT_REAL, CIN, 4, 4] f32 -> wpkT bf16 [(p*4+t)*COUTP + co][CIN]
__global__ void k_prep_convw(const float* w, u16* wpkT, int CIN, int COUT_REAL, int COUTP) {
    int i = blockIdx.x * blockDim.x + threadIdx.x;
    int total = 16 * COUTP * CIN;
    if (i >= total) return;
    int ci   = i % CIN;
    int rest = i / CIN;
    int co   = rest % COUTP;
    int pt   = rest / COUTP;
    int p = pt >> 2, t = pt & 3;
    int py = p >> 1, px = p & 1, ty = t >> 1, tx = t & 1;
    float v = 0.f;
    if (co < COUT_REAL) {
        int ky = tap_k(py, ty), kx = tap_k(px, tx);
        v = w[(((size_t)co * CIN + ci) * 4 + ky) * 4 + kx];
    }
    wpkT[i] = f2bf(v);
}

// ------------------------------------------------------------ Wigner apply
__global__ void k_wigner(const float* angles, const float* item_rep,
                         const float* jd, u16* item_bf, int N) {
    int gid = blockIdx.x * blockDim.x + threadIdx.x;
    if (gid >= N * 10) return;
    int n = gid / 10, r = gid % 10;
    float a0 = angles[n * 3 + 0], a1 = angles[n * 3 + 1], a2 = angles[n * 3 + 2];
    for (int l = 0; l <= 5; ++l) {
        int s = 2 * l + 1;
        const float* J = jd + JDOFF[l];
        float v[11], t[11];
        for (int i = 0; i < s; ++i) v[i] = item_rep[(l * l + i) * 10 + r];
        for (int i = 0; i < s; ++i) {
            float f = (float)(l - i);
            t[i] = __cosf(f * a2) * v[i] + __sinf(f * a2) * v[s - 1 - i];
        }
        for (int i = 0; i < s; ++i) {
            float acc = 0.f;
            for (int j = 0; j < s; ++j) acc += J[i * s + j] * t[j];
            v[i] = acc;
        }
        for (int i = 0; i < s; ++i) {
            float f = (float)(l - i);
            t[i] = __cosf(f * a1) * v[i] + __sinf(f * a1) * v[s - 1 - i];
        }
        for (int i = 0; i < s; ++i) {
            float acc = 0.f;
            for (int j = 0; j < s; ++j) acc += J[i * s + j] * t[j];
            v[i] = acc;
        }
        for (int i = 0; i < s; ++i) {
            float f = (float)(l - i);
            float o = __cosf(f * a0) * v[i] + __sinf(f * a0) * v[s - 1 - i];
            item_bf[(size_t)n * 384 + (l * l + i) * 10 + r] = f2bf(o);
        }
    }
}

// -------------------------------------------------- GEMM: relu(A@B + bias)
// Double-buffered LDS, async global->LDS staging (3 async ops per stage per
// wave: 1 A-chunk + 2 B-chunks), WMMA on the live buffer while next streams.
__global__ __launch_bounds__(256) void k_gemm_bias_relu(
    const u16* __restrict__ A, const u16* __restrict__ Bt,
    const float* __restrict__ bias, u16* __restrict__ out,
    int M, int Kp, int Nc) {
    __shared__ __align__(16) u16 lds_a[2][64 * 40];
    __shared__ __align__(16) u16 lds_b[2][128 * 40];
    const int tid = threadIdx.x, wave = tid >> 5, lane = tid & 31;
    const int mblk = blockIdx.y * 64, nblk = blockIdx.x * 128;
    const int mt = wave & 3, ntb = (wave >> 2) * 4;
    v8f acc[4]; for (int j = 0; j < 4; ++j) acc[j] = vzero8();

    const int arow = tid >> 2, apart = tid & 3;     // 64 rows x 4 chunks of 8
    const int brow = tid >> 1, bpart = tid & 1;     // 128 rows x 2 chunks of 16

    auto stage = [&](int buf, int kb) {
        async_ld_b128(A + (size_t)(mblk + arow) * Kp + kb + apart * 8,
                      &lds_a[buf][arow * 40 + apart * 8]);
        const u16* src = Bt + (size_t)(nblk + brow) * Kp + kb + bpart * 16;
        async_ld_b128(src,     &lds_b[buf][brow * 40 + bpart * 16]);
        async_ld_b128(src + 8, &lds_b[buf][brow * 40 + bpart * 16 + 8]);
    };

    stage(0, 0);
    int cur = 0;
    for (int kb = 0; kb < Kp; kb += 32) {
        const bool hn = (kb + 32) < Kp;
        if (hn) { stage(cur ^ 1, kb + 32); s_wait_async<3>(); }
        else    { s_wait_async<0>(); }
        __syncthreads();
        v16bf a = load_frag_a(lds_a[cur], mt, lane);
        for (int j = 0; j < 4; ++j) {
            v16bf b = load_frag_b(lds_b[cur], ntb + j, lane);
            acc[j] = wmma_bf16(a, b, acc[j]);
        }
        __syncthreads();
        cur ^= 1;
    }
    const int lo = lane & 15, hi = lane >> 4;
    for (int j = 0; j < 4; ++j) {
        int ncol = nblk + (ntb + j) * 16 + lo;
        float bv = bias[ncol];
        for (int r = 0; r < 8; ++r) {
            int mrow = mblk + mt * 16 + hi * 8 + r;
            float v = acc[j][r] + bv;
            v = v > 0.f ? v : 0.f;
            out[(size_t)mrow * Nc + ncol] = f2bf(v);
        }
    }
}

// act0 [n][j] (j = c*64+y*8+x) -> NHWC [n][y][x][c], coalesced writes
__global__ void k_relayout_nhwc(const u16* __restrict__ lin, u16* __restrict__ nhwc) {
    int i = blockIdx.x * blockDim.x + threadIdx.x;
    if (i >= 2048 * 16384) return;
    int n = i >> 14, rem = i & 16383;
    int pix = rem >> 8, c = rem & 255;
    int y = pix >> 3, x = pix & 7;
    nhwc[i] = lin[(size_t)n * 16384 + c * 64 + y * 8 + x];
}

// ------------------------------------------- fused conv-transpose via WMMA
// Parity-decomposed; A rows gathered from NHWC input (OOB lanes redirect to a
// zero-filled global buffer so async-op counts stay uniform per wave).
template <int CIN, int COUTP, int COUT_REAL, int HIN, int WIN, bool RELU, bool OUT_NCHW>
__global__ __launch_bounds__(256) void k_convt(
    const u16* __restrict__ in, const u16* __restrict__ wpkT,
    const u16* __restrict__ zerobuf,
    u16* __restrict__ out_bf, float* __restrict__ out_f32) {
    constexpr int HOUT = HIN * 2, WOUT = WIN * 2;
    constexpr int NTILES = COUTP / 16;
    constexpr int TILES  = 4 * NTILES;
    constexpr int TPW    = (TILES + 7) / 8;
    constexpr int BITERS = (COUTP * 4 + 255) / 256;   // B async ops per thread
    constexpr int PERW   = 1 + BITERS;                // async ops per stage

    __shared__ __align__(16) u16 lds_a[2][64 * 40];
    __shared__ __align__(16) u16 lds_b[2][COUTP * 40];

    const int tid = threadIdx.x, wave = tid >> 5, lane = tid & 31;
    const int n = blockIdx.z, p = blockIdx.y;
    const int py = p >> 1, px = p & 1;
    const int pixbase = blockIdx.x * 64;

    const int arow = tid >> 2, apart = tid & 3;
    const int abase = pixbase + arow;
    const int my = abase / WIN, mx = abase % WIN;

    v8f acc[TPW];
    for (int j = 0; j < TPW; ++j) acc[j] = vzero8();

    auto stage = [&](int buf, int kb) {
        const int t = kb / CIN, ci0 = kb % CIN;
        const int ty = t >> 1, tx = t & 1;
        const int iy = my + tap_di(py, ty);
        const int ix = mx + tap_di(px, tx);
        const bool valid = (iy >= 0) && (iy < HIN) && (ix >= 0) && (ix < WIN);
        const u16* gsrc = valid
            ? in + ((size_t)(n * HIN + iy) * WIN + ix) * CIN + ci0 + apart * 8
            : zerobuf;
        async_ld_b128(gsrc, &lds_a[buf][arow * 40 + apart * 8]);
        for (int it = 0; it < BITERS; ++it) {
            int i = (tid + it * 256) % (COUTP * 4);
            int brow = i >> 2, bpart = i & 3;
            async_ld_b128(wpkT + ((size_t)(p * 4 + t) * COUTP + brow) * CIN + ci0 + bpart * 8,
                          &lds_b[buf][brow * 40 + bpart * 8]);
        }
    };

    stage(0, 0);
    int cur = 0;
    for (int kb = 0; kb < 4 * CIN; kb += 32) {
        const bool hn = (kb + 32) < 4 * CIN;
        if (hn) { stage(cur ^ 1, kb + 32); s_wait_async<PERW>(); }
        else    { s_wait_async<0>(); }
        __syncthreads();
        for (int tt = 0; tt < TPW; ++tt) {
            int tile = wave + tt * 8;
            if (tile < TILES) {
                int mt = tile & 3, nt = tile >> 2;
                v16bf a = load_frag_a(lds_a[cur], mt, lane);
                v16bf b = load_frag_b(lds_b[cur], nt, lane);
                acc[tt] = wmma_bf16(a, b, acc[tt]);
            }
        }
        __syncthreads();
        cur ^= 1;
    }

    const int lo = lane & 15, hi = lane >> 4;
    for (int tt = 0; tt < TPW; ++tt) {
        int tile = wave + tt * 8;
        if (tile >= TILES) continue;
        int mt = tile & 3, nt = tile >> 2;
        int co = nt * 16 + lo;
        for (int r = 0; r < 8; ++r) {
            int m = pixbase + mt * 16 + hi * 8 + r;
            int oy = 2 * (m / WIN) + py, ox = 2 * (m % WIN) + px;
            float v = acc[tt][r];
            if (RELU) v = v > 0.f ? v : 0.f;
            if (OUT_NCHW) {
                if (co < COUT_REAL)
                    out_f32[((size_t)(n * COUT_REAL + co) * HOUT + oy) * WOUT + ox] = v;
            } else {
                out_bf[((size_t)(n * HOUT + oy) * WOUT + ox) * COUTP + co] = f2bf(v);
            }
        }
    }
}

// ------------------------------------------------------------------- host
extern "C" void kernel_launch(void* const* d_in, const int* in_sizes, int n_in,
                              void* d_out, int out_size, void* d_ws, size_t ws_size,
                              hipStream_t stream) {
    const float* angles   = (const float*)d_in[0];   // [2048,3]
    const float* item_rep = (const float*)d_in[1];   // [36,10]
    const float* w0       = (const float*)d_in[2];   // [360,16384]
    const float* b0       = (const float*)d_in[3];   // [16384]
    const float* w1       = (const float*)d_in[4];   // [128,256,4,4]
    const float* w2       = (const float*)d_in[5];   // [64,128,4,4]
    const float* w3       = (const float*)d_in[6];   // [3,64,4,4]
    (void)in_sizes; (void)n_in; (void)out_size; (void)ws_size;

    char* ws = (char*)d_ws;
    size_t off = 0;
    auto take = [&](size_t bytes) { size_t o = off; off = (off + bytes + 255) & ~(size_t)255; return o; };
    float* jd      = (float*)(ws + take(286 * 4));
    u16*   zerobuf = (u16*)  (ws + take(128 * 2));
    u16*   item_bf = (u16*)  (ws + take((size_t)2048 * 384 * 2));
    u16*   w0T     = (u16*)  (ws + take((size_t)16384 * 384 * 2));
    u16*   act0    = (u16*)  (ws + take((size_t)2048 * 16384 * 2));
    u16*   act0n   = (u16*)  (ws + take((size_t)2048 * 16384 * 2));
    u16*   w1p     = (u16*)  (ws + take((size_t)16 * 128 * 256 * 2));
    u16*   act1    = (u16*)  (ws + take((size_t)2048 * 16 * 16 * 128 * 2));
    u16*   w2p     = (u16*)  (ws + take((size_t)16 * 64 * 128 * 2));
    u16*   act2    = (u16*)  (ws + take((size_t)2048 * 32 * 32 * 64 * 2));
    u16*   w3p     = (u16*)  (ws + take((size_t)16 * 16 * 64 * 2));

    const int B = 256;
    // 1) Wigner J blocks + item in padded bf16
    k_init_jd<<<1, 1, 0, stream>>>(jd);
    k_fill_u16<<<1, 128, 0, stream>>>(zerobuf, 128, (u16)0);
    k_fill_u16<<<(2048 * 384 + B - 1) / B, B, 0, stream>>>(item_bf, 2048 * 384, (u16)0);
    k_wigner<<<(2048 * 10 + B - 1) / B, B, 0, stream>>>(angles, item_rep, jd, item_bf, 2048);
    // 2) weight prepack (bf16, padded, transposed for B-fragment locality)
    k_prep_w0<<<(16384 * 384 + B - 1) / B, B, 0, stream>>>(w0, w0T);
    k_prep_convw<<<(16 * 128 * 256 + B - 1) / B, B, 0, stream>>>(w1, w1p, 256, 128, 128);
    k_prep_convw<<<(16 * 64  * 128 + B - 1) / B, B, 0, stream>>>(w2, w2p, 128, 64, 64);
    k_prep_convw<<<(16 * 16  * 64  + B - 1) / B, B, 0, stream>>>(w3, w3p, 64, 3, 16);
    // 3) dense stem: relu(item @ w0 + b0) -> [2048,16384] bf16 -> NHWC
    k_gemm_bias_relu<<<dim3(16384 / 128, 2048 / 64), B, 0, stream>>>(
        item_bf, w0T, b0, act0, 2048, 384, 16384);
    k_relayout_nhwc<<<(2048 * 16384 + B - 1) / B, B, 0, stream>>>(act0, act0n);
    // 4) conv-transpose stack (parity-decomposed WMMA GEMMs, async staging)
    k_convt<256, 128, 128, 8, 8, true, false>
        <<<dim3(1, 4, 2048), B, 0, stream>>>(act0n, w1p, zerobuf, act1, nullptr);
    k_convt<128, 64, 64, 16, 16, true, false>
        <<<dim3(4, 4, 2048), B, 0, stream>>>(act1, w2p, zerobuf, act2, nullptr);
    k_convt<64, 16, 3, 32, 32, false, true>
        <<<dim3(16, 4, 2048), B, 0, stream>>>(act2, w3p, zerobuf, nullptr, (float*)d_out);
}
